// InverseDynamics_40664750359099
// MI455X (gfx1250) — matrix operand order repaired and checked by reading, not compiled, not run on hardware
//
#include <hip/hip_runtime.h>
#include <hip/hip_bf16.h>

typedef __bf16 bf16_t;
typedef __attribute__((ext_vector_type(16))) __bf16 v16bf;
typedef __attribute__((ext_vector_type(8)))  __bf16 v8bf;
typedef __attribute__((ext_vector_type(8)))  float   v8f;
typedef __attribute__((ext_vector_type(4)))  int     v4i;

#define DIMV   512
#define NCODES 16384   // A
#define NBATCH 2048    // B
#define NTOK   64      // H*W

#define AS_GLOBAL __attribute__((address_space(1)))
#define AS_LDS    __attribute__((address_space(3)))

#if defined(__has_builtin)
#  if __has_builtin(__builtin_amdgcn_global_load_async_to_lds_b128)
#    define ASYNC_LDS 1
#  endif
#endif
#ifndef ASYNC_LDS
#  define ASYNC_LDS 0
#endif

// ---------------------------------------------------------------------------
// contiguous global -> LDS tile copy (bytes multiple of 256*16), 256 threads
__device__ __forceinline__ void stage_tile(bf16_t* __restrict__ dst,
                                           const bf16_t* __restrict__ src,
                                           int bytes, int tid) {
#if ASYNC_LDS
    AS_GLOBAL v4i* g = (AS_GLOBAL v4i*)src;
    AS_LDS    v4i* l = (AS_LDS    v4i*)dst;
    const int n16 = bytes >> 4;
    for (int i = tid; i < n16; i += 256)
        __builtin_amdgcn_global_load_async_to_lds_b128(g + i, l + i, 0, 0);
#  if __has_builtin(__builtin_amdgcn_s_wait_asynccnt)
    __builtin_amdgcn_s_wait_asynccnt(0);
#  else
    asm volatile("s_wait_asynccnt 0" ::: "memory");
#  endif
#else
    const int nelems = bytes >> 1;
    for (int i = tid * 8; i < nelems; i += 256 * 8)
        *(v8bf*)&dst[i] = *(const v8bf*)&src[i];
#endif
}

// order-preserving float -> uint32 (total order, ascending)
__device__ __forceinline__ unsigned int fflip(float x) {
    unsigned int u = __float_as_uint(x);
    return (u & 0x80000000u) ? ~u : (u | 0x80000000u);
}

__device__ __forceinline__ v8f wmma_bf16(v16bf a, v16bf b, v8f c) {
    return __builtin_amdgcn_wmma_f32_16x16x32_bf16(false, a, false, b,
                                                   (short)0, c, false, false);
}

// ---------------------------------------------------------------------------
// Register-blocked (MR x 4) x K MAC loop with register double-buffering.
//  abase: LDS, pre-offset &Atile[row0*KDIM + lhalf*16]; M-tile r at +r*16*KDIM
//  bbase: global, pre-offset Bt + (ncol0+lmod)*KDIM;   N-tile t at +t*16*KDIM
// All addressing folds into immediate offsets off two base pointers.
template <int KDIM, int MR>
__device__ __forceinline__ void mac_loop(const bf16_t* __restrict__ abase,
                                         const bf16_t* __restrict__ bbase,
                                         int lhalf, v8f acc[MR][4]) {
    v8bf alo[MR], ahi[MR];
    v16bf bcur[4];
#pragma unroll
    for (int r = 0; r < MR; r++) {
        alo[r] = *(const v8bf*)(abase + r * 16 * KDIM);
        ahi[r] = *(const v8bf*)(abase + r * 16 * KDIM + 8);
    }
#pragma unroll
    for (int t = 0; t < 4; t++)
        bcur[t] = *(const v16bf*)(bbase + t * 16 * KDIM + lhalf * 16);

#pragma unroll
    for (int k = 0; k < KDIM; k += 32) {
        v16bf afrag[MR];
#pragma unroll
        for (int r = 0; r < MR; r++)
#pragma unroll
            for (int i = 0; i < 8; i++) {
                afrag[r][i] = alo[r][i]; afrag[r][i + 8] = ahi[r][i];
            }
        v16bf bnxt[4];
        if (k + 32 < KDIM) {       // prefetch next step into fresh registers
#pragma unroll
            for (int r = 0; r < MR; r++) {
                alo[r] = *(const v8bf*)(abase + r * 16 * KDIM + k + 32);
                ahi[r] = *(const v8bf*)(abase + r * 16 * KDIM + k + 40);
            }
#pragma unroll
            for (int t = 0; t < 4; t++)
                bnxt[t] = *(const v16bf*)(bbase + t * 16 * KDIM + k + 32 + lhalf * 16);
        }
#pragma unroll
        for (int t = 0; t < 4; t++)
#pragma unroll
            for (int r = 0; r < MR; r++)
                acc[r][t] = wmma_bf16(afrag[r], bcur[t], acc[r][t]);
#pragma unroll
        for (int t = 0; t < 4; t++) bcur[t] = bnxt[t];
    }
}

// ---------------------------------------------------------------------------
__global__ void init_best(unsigned long long* best) {
    int b = blockIdx.x * blockDim.x + threadIdx.x;
    if (b < NBATCH) best[b] = ~0ull;
}

// wt[n][k] = bf16(w[k][n])   (w is [Kdim, Ndim] row-major)
__global__ void prep_wt(const float* __restrict__ w, bf16_t* __restrict__ wt,
                        int Kdim, int Ndim) {
    int i = blockIdx.x * blockDim.x + threadIdx.x;
    if (i < Kdim * Ndim) {
        int n = i / Kdim, k = i % Kdim;
        wt[i] = (bf16_t)w[(size_t)k * Ndim + n];
    }
}

// codebook fp32 -> bf16 + per-row squared norm
__global__ __launch_bounds__(256) void prep_cb(const float* __restrict__ cb,
                                               bf16_t* __restrict__ cbb,
                                               float* __restrict__ cnorm) {
    int a = blockIdx.x;
    int t = threadIdx.x;
    float s = 0.f;
    for (int c = t; c < DIMV; c += 256) {
        float v = cb[(size_t)a * DIMV + c];
        cbb[(size_t)a * DIMV + c] = (bf16_t)v;
        s += v * v;
    }
    for (int off = 16; off > 0; off >>= 1) s += __shfl_xor(s, off, 32);
    __shared__ float red[8];
    if ((t & 31) == 0) red[t >> 5] = s;
    __syncthreads();
    if (t == 0) {
        float tot = 0.f;
        for (int i = 0; i < 8; i++) tot += red[i];
        cnorm[a] = tot;
    }
}

// ---------------------------------------------------------------------------
// embedding gather + mean pool -> h[b, 0:512]=mean(e_t), h[b,512:1024]=mean(e_tp1)
__global__ __launch_bounds__(256) void embed_pool(const int* __restrict__ zt,
                                                  const int* __restrict__ ztp1,
                                                  const float* __restrict__ emb,
                                                  bf16_t* __restrict__ h) {
    int b = blockIdx.x;
    __shared__ int zidx[2 * NTOK];
    if (threadIdx.x < NTOK)            zidx[threadIdx.x] = zt[b * NTOK + threadIdx.x];
    else if (threadIdx.x < 2 * NTOK)   zidx[threadIdx.x] = ztp1[b * NTOK + (threadIdx.x - NTOK)];
    __syncthreads();
    for (int d = threadIdx.x; d < 2 * DIMV; d += 256) {
        int half = d >> 9;
        int ch = d & (DIMV - 1);
        const int* z = &zidx[half * NTOK];
        float s = 0.f;
        for (int t = 0; t < NTOK; t++)
            s += emb[(size_t)z[t] * DIMV + ch];
        h[(size_t)b * (2 * DIMV) + d] = (bf16_t)(s * (1.0f / NTOK));
    }
}

// ---------------------------------------------------------------------------
// C[m,n] = act(A @ Bt^T + bias); block = 32 rows x 512 cols,
// every wave covers 2 M-tiles x 4 N-tiles (wave w owns cols w*64..w*64+63)
template <int KDIM, bool DO_GELU>
__global__ __launch_bounds__(256) void gemm_bias_act(const bf16_t* __restrict__ A,
                                                     const bf16_t* __restrict__ Bt,
                                                     const float* __restrict__ bias,
                                                     bf16_t* __restrict__ C) {
    __shared__ __align__(16) bf16_t Atile[32 * KDIM];
    const int m0 = blockIdx.x * 32;
    const int tid = threadIdx.x;

    stage_tile(Atile, A + (size_t)m0 * KDIM, 32 * KDIM * 2, tid);
    __syncthreads();

    const int wave = tid >> 5, lane = tid & 31;
    const int lhalf = lane >> 4, lmod = lane & 15;

    v8f acc[2][4];
#pragma unroll
    for (int r = 0; r < 2; r++)
#pragma unroll
        for (int t = 0; t < 4; t++) acc[r][t] = (v8f){0.f,0.f,0.f,0.f,0.f,0.f,0.f,0.f};

    const bf16_t* abase = &Atile[lmod * KDIM + lhalf * 16];
    const bf16_t* bbase = Bt + (size_t)(wave * 64 + lmod) * KDIM;
    mac_loop<KDIM, 2>(abase, bbase, lhalf, acc);

#pragma unroll
    for (int t = 0; t < 4; t++) {
        int ncol = wave * 64 + t * 16 + lmod;
        float bv = bias[ncol];
#pragma unroll
        for (int r = 0; r < 2; r++)
#pragma unroll
            for (int j = 0; j < 8; j++) {
                float x = acc[r][t][j] + bv;
                if (DO_GELU) x = 0.5f * x * (1.0f + erff(x * 0.70710678f));
                int m = m0 + r * 16 + lhalf * 8 + j;
                C[(size_t)m * DIMV + ncol] = (bf16_t)x;
            }
    }
}

// ---------------------------------------------------------------------------
// VQ: score[m,n] = ||c_n||^2 - 2 <e_m, c_n>; block = 64 rows x 256 cols,
// waves arranged 2(M-groups) x 4(N); each wave = 2 M-tiles x 4 N-tiles.
__global__ __launch_bounds__(256) void vq_argmin(const bf16_t* __restrict__ e,
                                                 const bf16_t* __restrict__ cb,
                                                 const float* __restrict__ cnorm,
                                                 unsigned long long* __restrict__ best) {
    __shared__ __align__(16) bf16_t Atile[64 * DIMV];   // 64 KB
    __shared__ unsigned long long lmin[64];
    const int m0 = blockIdx.x * 64;
    const int n0 = blockIdx.y * 256;
    const int tid = threadIdx.x;

    stage_tile(Atile, e + (size_t)m0 * DIMV, 64 * DIMV * 2, tid);
    if (tid < 64) lmin[tid] = ~0ull;
    __syncthreads();

    const int wave = tid >> 5, lane = tid & 31;
    const int wm = wave >> 2, wn = wave & 3;      // wm: rows wm*32..wm*32+31
    const int lhalf = lane >> 4, lmod = lane & 15;

    v8f acc[2][4];
#pragma unroll
    for (int r = 0; r < 2; r++)
#pragma unroll
        for (int t = 0; t < 4; t++) acc[r][t] = (v8f){0.f,0.f,0.f,0.f,0.f,0.f,0.f,0.f};

    const bf16_t* abase = &Atile[(wm * 32 + lmod) * DIMV + lhalf * 16];
    const bf16_t* bbase = cb + (size_t)(n0 + wn * 64 + lmod) * DIMV;
    mac_loop<DIMV, 2>(abase, bbase, lhalf, acc);

    float cn[4];
    int ncol[4];
#pragma unroll
    for (int t = 0; t < 4; t++) {
        ncol[t] = n0 + wn * 64 + t * 16 + lmod;
        cn[t] = cnorm[ncol[t]];
    }

#pragma unroll
    for (int r = 0; r < 2; r++)
#pragma unroll
        for (int j = 0; j < 8; j++) {
            unsigned long long key = ~0ull;
#pragma unroll
            for (int t = 0; t < 4; t++) {
                float score = cn[t] - 2.0f * acc[r][t][j];
                unsigned long long kk =
                    ((unsigned long long)fflip(score) << 32) | (unsigned int)ncol[t];
                key = (kk < key) ? kk : key;
            }
            // min across the 16 lanes of this half-wave (tile columns)
            for (int off = 1; off < 16; off <<= 1) {
                unsigned long long o = __shfl_xor(key, off, 32);
                key = (o < key) ? o : key;
            }
            if (lmod == 0)
                atomicMin(&lmin[wm * 32 + r * 16 + lhalf * 8 + j], key);
        }
    __syncthreads();
    if (tid < 64) atomicMin(&best[m0 + tid], lmin[tid]);
}

// ---------------------------------------------------------------------------
__global__ __launch_bounds__(128) void finalize(const unsigned long long* __restrict__ best,
                                                const float* __restrict__ cb,
                                                float* __restrict__ out_ids,
                                                float* __restrict__ out_q) {
    int b = blockIdx.x;
    unsigned int id = (unsigned int)(best[b] & 0xFFFFFFFFu);
    if (threadIdx.x == 0) out_ids[b] = (float)id;
    for (int c = threadIdx.x; c < DIMV; c += 128)
        out_q[(size_t)b * DIMV + c] = cb[(size_t)id * DIMV + c];
}

// ---------------------------------------------------------------------------
extern "C" void kernel_launch(void* const* d_in, const int* in_sizes, int n_in,
                              void* d_out, int out_size, void* d_ws, size_t ws_size,
                              hipStream_t stream) {
    const int*   z_t   = (const int*)d_in[0];
    const int*   z_tp1 = (const int*)d_in[1];
    const float* emb   = (const float*)d_in[2];
    const float* w1    = (const float*)d_in[3];
    const float* b1    = (const float*)d_in[4];
    const float* w2    = (const float*)d_in[5];
    const float* b2    = (const float*)d_in[6];
    const float* cbf   = (const float*)d_in[7];

    char* ws = (char*)d_ws;
    bf16_t* h    = (bf16_t*)ws;                 ws += (size_t)NBATCH * 1024 * 2;  // 4 MB
    bf16_t* w1t  = (bf16_t*)ws;                 ws += (size_t)512 * 1024 * 2;     // 1 MB
    bf16_t* h2   = (bf16_t*)ws;                 ws += (size_t)NBATCH * 512 * 2;   // 2 MB
    bf16_t* w2t  = (bf16_t*)ws;                 ws += (size_t)512 * 512 * 2;      // 0.5 MB
    bf16_t* ebf  = (bf16_t*)ws;                 ws += (size_t)NBATCH * 512 * 2;   // 2 MB
    bf16_t* cbb  = (bf16_t*)ws;                 ws += (size_t)NCODES * 512 * 2;   // 16 MB
    float*  cnrm = (float*)ws;                  ws += (size_t)NCODES * 4;         // 64 KB
    unsigned long long* best = (unsigned long long*)ws;                           // 16 KB

    float* out_ids = (float*)d_out;
    float* out_q   = (float*)d_out + NBATCH;

    // prep
    init_best<<<(NBATCH + 255) / 256, 256, 0, stream>>>(best);
    prep_wt<<<(1024 * 512 + 255) / 256, 256, 0, stream>>>(w1, w1t, 1024, 512);
    prep_wt<<<(512 * 512 + 255) / 256, 256, 0, stream>>>(w2, w2t, 512, 512);
    prep_cb<<<NCODES, 256, 0, stream>>>(cbf, cbb, cnrm);

    // embed + pool
    embed_pool<<<NBATCH, 256, 0, stream>>>(z_t, z_tp1, emb, h);

    // MLP (each block covers 32 rows x full N=512)
    gemm_bias_act<1024, true><<<NBATCH / 32, 256, 0, stream>>>(h, w1t, b1, h2);
    gemm_bias_act<512, false><<<NBATCH / 32, 256, 0, stream>>>(h2, w2t, b2, ebf);

    // VQ distance + fused argmin (64 rows x 256 codes per block)
    vq_argmin<<<dim3(NBATCH / 64, NCODES / 256), 256, 0, stream>>>(ebf, cbb, cnrm, best);

    // gather winners
    finalize<<<NBATCH, 128, 0, stream>>>(best, cbf, out_ids, out_q);
}
